// Seq2SeqRNN_attn_8701603741865
// MI455X (gfx1250) — matrix operand order repaired and checked by reading, not compiled, not run on hardware
//
#include <hip/hip_runtime.h>
#include <hip/hip_bf16.h>

// ---------------------------------------------------------------------------
// Types for CDNA5 WMMA (wave32, v_wmma_f32_16x16x32_bf16)
// ---------------------------------------------------------------------------
typedef unsigned short ushort_t;
typedef __bf16 bf16_t;
typedef bf16_t bf16x16 __attribute__((ext_vector_type(16)));
typedef float  floatx8 __attribute__((ext_vector_type(8)));
typedef int    intx4   __attribute__((ext_vector_type(4)));

union FragBF { bf16x16 v; intx4 q[2]; };

__device__ __forceinline__ ushort_t f2bf(float f) {
    union { float f; unsigned u; } v; v.f = f;
    unsigned r = v.u + 0x7FFFu + ((v.u >> 16) & 1u);   // round-to-nearest-even
    return (ushort_t)(r >> 16);
}

// ---------------------------------------------------------------------------
// bf16 WMMA GEMM:  C(M,N) = A(M,K) * B(N,K)^T + bias
//   Requirements (enforced by host): N % 64 == 0, K % 32 == 0, M % 16 == 0,
//   bias != nullptr (point at a zeroed buffer when no bias; biasN clamps).
//   Each wave computes a 16x64 strip: 1 A-fragment reused by 4 WMMAs.
//   The body is straight-line (no branches after the top early-exit) so the
//   4x8-VGPR accumulators stay pinned and EXEC is all-ones around WMMA.
//   Store offset = (m % row_mod)*stride_r + (m / row_mod)*stride_q + n
// ---------------------------------------------------------------------------
__global__ void __launch_bounds__(256)
wmma_gemm_bf16(const ushort_t* __restrict__ A, int lda,
               const ushort_t* __restrict__ B, int ldb,
               const float* __restrict__ bias, int biasN,
               float* __restrict__ C,
               int M, int N, int K,
               int row_mod, long long stride_r, long long stride_q)
{
    const int lane = threadIdx.x & 31;
    // wave-uniform tile id, proven scalar to the compiler
    const int wid = __builtin_amdgcn_readfirstlane(
        (int)((blockIdx.x * blockDim.x + threadIdx.x) >> 5));
    const int ngroups = N >> 6;                     // N is a multiple of 64
    const int mt = wid / ngroups;
    if (mt * 16 >= M) return;                       // scalar branch (early out)
    const int ng = wid - mt * ngroups;
    const int m0 = mt * 16;
    const int n0base = ng * 64;

    const int halfsel = lane >> 4;                  // 0: lanes 0-15, 1: lanes 16-31
    const int rA = lane & 15;

    // A fragment per ISA 7.12.2: lanes 0-15 hold K {0..7,16..23},
    // lanes 16-31 hold K {8..15,24..31}
    const ushort_t* Arow = A + (size_t)(m0 + rA) * lda + (size_t)(halfsel * 8);
    // B fragment: column n = n0+rA is row (n0+rA) of W; lanes 0-15 carry
    // K {0..7,16..23}, lanes 16-31 carry K {8..15,24..31} (2 x b128)
    const ushort_t* Bbase = B + (size_t)(n0base + rA) * ldb + (size_t)(halfsel * 8);

    floatx8 acc[4];
#pragma unroll
    for (int j = 0; j < 4; ++j) {
        int n = n0base + j * 16 + rA;
        int nc = (n < biasN) ? n : 0;                // always-in-bounds load
        float bv = bias[nc];
        if (n >= biasN) bv = 0.0f;                   // lane select, no branch
#pragma unroll
        for (int r = 0; r < 8; ++r) acc[j][r] = bv;
    }

    for (int k0 = 0; k0 < K; k0 += 32) {
        FragBF a;
        a.q[0] = *reinterpret_cast<const intx4*>(Arow + k0);
        a.q[1] = *reinterpret_cast<const intx4*>(Arow + k0 + 16);
        __builtin_prefetch((const void*)(Arow + k0 + 32), 0, 0); // speculative
        FragBF b0, b1, b2, b3;
        {
            const ushort_t* Br = Bbase + k0;
            b0.q[0] = *reinterpret_cast<const intx4*>(Br);
            b0.q[1] = *reinterpret_cast<const intx4*>(Br + 16);
            Br += (size_t)16 * ldb;
            b1.q[0] = *reinterpret_cast<const intx4*>(Br);
            b1.q[1] = *reinterpret_cast<const intx4*>(Br + 16);
            Br += (size_t)16 * ldb;
            b2.q[0] = *reinterpret_cast<const intx4*>(Br);
            b2.q[1] = *reinterpret_cast<const intx4*>(Br + 16);
            Br += (size_t)16 * ldb;
            b3.q[0] = *reinterpret_cast<const intx4*>(Br);
            b3.q[1] = *reinterpret_cast<const intx4*>(Br + 16);
        }
        acc[0] = __builtin_amdgcn_wmma_f32_16x16x32_bf16(
            false, a.v, false, b0.v, (short)0, acc[0], false, false);
        acc[1] = __builtin_amdgcn_wmma_f32_16x16x32_bf16(
            false, a.v, false, b1.v, (short)0, acc[1], false, false);
        acc[2] = __builtin_amdgcn_wmma_f32_16x16x32_bf16(
            false, a.v, false, b2.v, (short)0, acc[2], false, false);
        acc[3] = __builtin_amdgcn_wmma_f32_16x16x32_bf16(
            false, a.v, false, b3.v, (short)0, acc[3], false, false);
    }

    // C/D layout: VGPR r -> M = m0 + r + halfsel*8, N = n0 + (lane&15)
#pragma unroll
    for (int j = 0; j < 4; ++j) {
        int n = n0base + j * 16 + rA;
#pragma unroll
        for (int r = 0; r < 8; ++r) {
            int m = m0 + r + halfsel * 8;
            long long o = (long long)(m % row_mod) * stride_r
                        + (long long)(m / row_mod) * stride_q + n;
            C[o] = acc[j][r];
        }
    }
}

// ---------------------------------------------------------------------------
// f32 -> bf16 with zero padding of extra rows/cols
// ---------------------------------------------------------------------------
__global__ void cvt_pad_bf16(const float* __restrict__ src, int sr, int sc, int sld,
                             ushort_t* __restrict__ dst, int dr, int dc)
{
    long long i = (long long)blockIdx.x * blockDim.x + threadIdx.x;
    long long total = (long long)dr * dc;
    if (i >= total) return;
    int r = (int)(i / dc), c = (int)(i % dc);
    float v = (r < sr && c < sc) ? src[(size_t)r * sld + c] : 0.0f;
    dst[i] = f2bf(v);
}

// ---------------------------------------------------------------------------
// Encoder embedding gather -> time-major bf16 X0[(t*64+b)*256 + e]
// ---------------------------------------------------------------------------
__global__ void embed_enc(const int* __restrict__ inp, const float* __restrict__ emb,
                          ushort_t* __restrict__ X0)
{
    int i = blockIdx.x * blockDim.x + threadIdx.x;
    if (i >= 50 * 64 * 256) return;
    int e = i & 255, rb = i >> 8, b = rb & 63, t = rb >> 6;
    int tok = inp[b * 50 + t];
    X0[i] = f2bf(emb[(size_t)tok * 256 + e]);
}

// ---------------------------------------------------------------------------
// Fused GRU gates (PyTorch order r,z,n), f32 math; updates h in place and
// emits bf16 copy (zero-padded to ldhbf) + optional duplicate outputs.
// ---------------------------------------------------------------------------
__global__ void gru_fuse(const float* __restrict__ gi, int ldgi,
                         const float* __restrict__ gh, int ldgh,
                         float* __restrict__ h, int ldh,
                         ushort_t* __restrict__ hbf, int ldhbf,
                         ushort_t* __restrict__ ybf, int ldy,
                         float* __restrict__ yf, int ldyf,
                         int H, int rows)
{
    int i = blockIdx.x * blockDim.x + threadIdx.x;
    if (i >= rows * ldhbf) return;
    int j = i % ldhbf, b = i / ldhbf;
    if (j >= H) { hbf[(size_t)b * ldhbf + j] = 0; return; }
    float ir = gi[(size_t)b * ldgi + j];
    float iz = gi[(size_t)b * ldgi + H + j];
    float ig = gi[(size_t)b * ldgi + 2 * H + j];
    float hr = gh[(size_t)b * ldgh + j];
    float hz = gh[(size_t)b * ldgh + H + j];
    float hg = gh[(size_t)b * ldgh + 2 * H + j];
    float r = 1.0f / (1.0f + __expf(-(ir + hr)));
    float z = 1.0f / (1.0f + __expf(-(iz + hz)));
    float n = tanhf(ig + r * hg);
    float hn = (1.0f - z) * n + z * h[(size_t)b * ldh + j];
    h[(size_t)b * ldh + j] = hn;
    hbf[(size_t)b * ldhbf + j] = f2bf(hn);
    if (ybf) ybf[(size_t)b * ldy + j] = f2bf(hn);
    if (yf)  yf[(size_t)b * ldyf + j] = hn;
}

// ---------------------------------------------------------------------------
// pre[0,b] = concat(hf0,hf1), pre[1,b] = concat(hb0,hb1)  (bf16, 128x512)
// ---------------------------------------------------------------------------
__global__ void build_pre(const float* __restrict__ h00, const float* __restrict__ h01,
                          const float* __restrict__ h10, const float* __restrict__ h11,
                          ushort_t* __restrict__ pre)
{
    int i = blockIdx.x * blockDim.x + threadIdx.x;
    if (i >= 128 * 512) return;
    int c = i & 511, rb = i >> 9, b = rb & 63, l = rb >> 6;
    const float* src = (c < 256) ? (l == 0 ? h00 : h01) : (l == 0 ? h10 : h11);
    pre[i] = f2bf(src[b * 256 + (c & 255)]);
}

// ---------------------------------------------------------------------------
// Attention scores: s[b,t] = sum_d tanh(enc_att[t*64+b,d] + hid_att[b,d]) * V[d]
// One wave per (b,t) row, wave32 shuffle reduction.
// ---------------------------------------------------------------------------
__global__ void attn_score(const float* __restrict__ enc_att,
                           const float* __restrict__ hid_att,
                           const float* __restrict__ V,
                           float* __restrict__ scores)
{
    int wid = (int)((blockIdx.x * blockDim.x + threadIdx.x) >> 5);
    int lane = threadIdx.x & 31;
    if (wid >= 3200) return;
    int b = wid & 63, t = wid >> 6;
    const float* ea = enc_att + (size_t)(t * 64 + b) * 320;
    const float* ha = hid_att + (size_t)b * 320;
    float s = 0.0f;
    for (int d = lane; d < 300; d += 32)
        s += tanhf(ea[d] + ha[d]) * V[d];
    for (int o = 16; o > 0; o >>= 1) s += __shfl_down(s, o, 32);
    if (lane == 0) scores[b * 50 + t] = s;
}

// ---------------------------------------------------------------------------
// Per-batch softmax over SL, context vector, and decoder-input assembly:
// x_bf[b, 0:300]   = emb_dec[tok]
// x_bf[b, 300:812] = sum_s w[s] * enc_out[s*64+b, :]
// x_bf[b, 812:832] = 0 (K padding)
// ---------------------------------------------------------------------------
__global__ void softmax_ctx_x(const float* __restrict__ scores,
                              const float* __restrict__ encout,
                              const float* __restrict__ emb_dec,
                              const int* __restrict__ targ, int t,
                              ushort_t* __restrict__ xbf)
{
    __shared__ float w[50];
    int b = blockIdx.x;
    if (threadIdx.x == 0) {
        float mx = -1e30f;
        for (int s = 0; s < 50; ++s) mx = fmaxf(mx, scores[b * 50 + s]);
        float sum = 0.0f;
        for (int s = 0; s < 50; ++s) { float e = __expf(scores[b * 50 + s] - mx); w[s] = e; sum += e; }
        float inv = 1.0f / sum;
        for (int s = 0; s < 50; ++s) w[s] *= inv;
    }
    __syncthreads();
    int tok = (t == 0) ? 0 : targ[b * 32 + (t - 1)];
    for (int c = threadIdx.x; c < 832; c += blockDim.x) {
        float v;
        if (c < 300) {
            v = emb_dec[(size_t)tok * 300 + c];
        } else if (c < 812) {
            int d = c - 300;
            float acc = 0.0f;
            for (int s = 0; s < 50; ++s)
                acc += w[s] * encout[((size_t)s * 64 + b) * 512 + d];
            v = acc;
        } else v = 0.0f;
        xbf[(size_t)b * 832 + c] = f2bf(v);
    }
}

// ---------------------------------------------------------------------------
// Host side
// ---------------------------------------------------------------------------
static void launch_gemm(hipStream_t s, const ushort_t* A, int lda,
                        const ushort_t* B, int ldb,
                        const float* bias, int biasN, float* C,
                        int M, int N, int K,
                        int row_mod, long long sr, long long sq)
{
    int ngroups = N / 64;
    long long waves = (long long)(M / 16) * ngroups;
    int blocks = (int)((waves + 7) / 8);
    wmma_gemm_bf16<<<blocks, 256, 0, s>>>(A, lda, B, ldb, bias, biasN, C,
                                          M, N, K, row_mod, sr, sq);
}

extern "C" void kernel_launch(void* const* d_in, const int* in_sizes, int n_in,
                              void* d_out, int out_size, void* d_ws, size_t ws_size,
                              hipStream_t stream)
{
    const int*   inp       = (const int*)d_in[0];
    const int*   targ      = (const int*)d_in[1];
    const float* emb_enc_w = (const float*)d_in[2];
    const float* emb_dec_w = (const float*)d_in[3];
    const float* e_wih0    = (const float*)d_in[4];
    const float* e_whh0    = (const float*)d_in[5];
    const float* e_bih0    = (const float*)d_in[6];
    const float* e_bhh0    = (const float*)d_in[7];
    const float* e_wih1    = (const float*)d_in[8];
    const float* e_whh1    = (const float*)d_in[9];
    const float* e_bih1    = (const float*)d_in[10];
    const float* e_bhh1    = (const float*)d_in[11];
    const float* out_enc_w = (const float*)d_in[12];
    const float* enc_att_w = (const float*)d_in[13];
    const float* hid_att_w = (const float*)d_in[14];
    const float* hid_att_b = (const float*)d_in[15];
    const float* Vvec      = (const float*)d_in[16];
    const float* dw_ih0    = (const float*)d_in[17];
    const float* dw_hh0    = (const float*)d_in[18];
    const float* db_ih0    = (const float*)d_in[19];
    const float* db_hh0    = (const float*)d_in[20];
    const float* dw_ih1    = (const float*)d_in[21];
    const float* dw_hh1    = (const float*)d_in[22];
    const float* db_ih1    = (const float*)d_in[23];
    const float* db_hh1    = (const float*)d_in[24];
    const float* out_w     = (const float*)d_in[25];
    const float* out_b     = (const float*)d_in[26];
    float* out = (float*)d_out;
    (void)in_sizes; (void)n_in; (void)out_size; (void)ws_size;

    // ---- workspace carve-up (256B aligned) ----
    // Padded dims: ED 300->320 (K mult 32), 3*ED 900->960 (N mult 64),
    // ED+2NH 812->832 (K mult 32).
    char* ws = (char*)d_ws; size_t off = 0;
    auto alloc = [&](size_t bytes) -> void* {
        void* p = ws + off; off += bytes; off = (off + 255) & ~(size_t)255; return p;
    };
    ushort_t *wih0_bf[2], *whh0_bf[2], *wih1_bf[2], *whh1_bf[2];
    for (int d = 0; d < 2; ++d) {
        wih0_bf[d] = (ushort_t*)alloc((size_t)768 * 256 * 2);
        whh0_bf[d] = (ushort_t*)alloc((size_t)768 * 256 * 2);
        wih1_bf[d] = (ushort_t*)alloc((size_t)768 * 512 * 2);
        whh1_bf[d] = (ushort_t*)alloc((size_t)768 * 256 * 2);
    }
    ushort_t* oenc_bf  = (ushort_t*)alloc((size_t)320 * 512 * 2);
    ushort_t* eattw_bf = (ushort_t*)alloc((size_t)320 * 512 * 2);
    ushort_t* hattw_bf = (ushort_t*)alloc((size_t)320 * 320 * 2);
    ushort_t* dwih0_bf = (ushort_t*)alloc((size_t)960 * 832 * 2);
    ushort_t* dwhh0_bf = (ushort_t*)alloc((size_t)960 * 320 * 2);
    ushort_t* dwih1_bf = (ushort_t*)alloc((size_t)960 * 320 * 2);
    ushort_t* dwhh1_bf = (ushort_t*)alloc((size_t)960 * 320 * 2);
    ushort_t* outw_bf  = (ushort_t*)alloc((size_t)32000 * 320 * 2);

    ushort_t* X0       = (ushort_t*)alloc((size_t)3200 * 256 * 2);
    ushort_t* xs1      = (ushort_t*)alloc((size_t)3200 * 512 * 2);
    ushort_t* encout_bf= (ushort_t*)alloc((size_t)3200 * 512 * 2);
    ushort_t* hbf_buf  = (ushort_t*)alloc((size_t)2 * 64 * 256 * 2);
    ushort_t* pre_bf   = (ushort_t*)alloc((size_t)128 * 512 * 2);
    ushort_t* hid_bf   = (ushort_t*)alloc((size_t)128 * 320 * 2);
    ushort_t* h0bf     = (ushort_t*)alloc((size_t)64 * 320 * 2);
    ushort_t* H1all    = (ushort_t*)alloc((size_t)2048 * 320 * 2);
    ushort_t* xbf      = (ushort_t*)alloc((size_t)64 * 832 * 2);

    float* gi[2];
    gi[0] = (float*)alloc((size_t)3200 * 768 * 4);
    gi[1] = (float*)alloc((size_t)3200 * 768 * 4);
    float* gh[2];
    gh[0] = (float*)alloc((size_t)64 * 768 * 4);
    gh[1] = (float*)alloc((size_t)64 * 768 * 4);
    float* hf32    = (float*)alloc((size_t)4 * 64 * 256 * 4);   // [layer*2+dir]
    float* encoutf = (float*)alloc((size_t)3200 * 512 * 4);
    float* encatt  = (float*)alloc((size_t)3200 * 320 * 4);
    float* hidf    = (float*)alloc((size_t)128 * 320 * 4);      // decoder h0,h1 f32
    float* hidatt  = (float*)alloc((size_t)64 * 320 * 4);
    float* scores  = (float*)alloc((size_t)64 * 50 * 4);
    float* gid     = (float*)alloc((size_t)64 * 960 * 4);
    float* ghd     = (float*)alloc((size_t)64 * 960 * 4);
    float* zbias   = (float*)alloc((size_t)320 * 4);            // zeroed "no bias"

    auto cvt = [&](const float* src, int sr, int sc, int sld, ushort_t* dst, int dr, int dc) {
        long long total = (long long)dr * dc;
        int blocks = (int)((total + 255) / 256);
        cvt_pad_bf16<<<blocks, 256, 0, stream>>>(src, sr, sc, sld, dst, dr, dc);
    };
    auto fuse = [&](const float* gip, int ldgi, const float* ghp, int ldgh,
                    float* h, int ldh, ushort_t* hbf, int ldhbf,
                    ushort_t* ybf, int ldy, float* yf, int ldyf, int H) {
        int total = 64 * ldhbf;
        gru_fuse<<<(total + 255) / 256, 256, 0, stream>>>(gip, ldgi, ghp, ldgh,
                                                          h, ldh, hbf, ldhbf,
                                                          ybf, ldy, yf, ldyf, H, 64);
    };
    const int BIGMOD = 1 << 30;

    hipMemsetAsync(zbias, 0, (size_t)320 * 4, stream);

    // ---- weight conversion f32 -> zero-padded bf16 ----
    for (int d = 0; d < 2; ++d) {
        cvt(e_wih0 + (size_t)d * 768 * 256, 768, 256, 256, wih0_bf[d], 768, 256);
        cvt(e_whh0 + (size_t)d * 768 * 256, 768, 256, 256, whh0_bf[d], 768, 256);
        cvt(e_wih1 + (size_t)d * 768 * 512, 768, 512, 512, wih1_bf[d], 768, 512);
        cvt(e_whh1 + (size_t)d * 768 * 256, 768, 256, 256, whh1_bf[d], 768, 256);
    }
    cvt(out_enc_w, 300, 512, 512, oenc_bf, 320, 512);
    cvt(enc_att_w, 300, 512, 512, eattw_bf, 320, 512);
    cvt(hid_att_w, 300, 300, 300, hattw_bf, 320, 320);
    cvt(dw_ih0, 900, 812, 812, dwih0_bf, 960, 832);
    cvt(dw_hh0, 900, 300, 300, dwhh0_bf, 960, 320);
    cvt(dw_ih1, 900, 300, 300, dwih1_bf, 960, 320);
    cvt(dw_hh1, 900, 300, 300, dwhh1_bf, 960, 320);
    cvt(out_w, 32000, 300, 300, outw_bf, 32000, 320);

    // ---- encoder ----
    embed_enc<<<(3200 * 256) / 256, 256, 0, stream>>>(inp, emb_enc_w, X0);
    hipMemsetAsync(hf32, 0, (size_t)4 * 64 * 256 * 4, stream);
    hipMemsetAsync(hbf_buf, 0, (size_t)2 * 64 * 256 * 2, stream);

    // layer 0: hoisted input GEMMs (3200x768x256), then 50 recurrent steps/dir
    for (int d = 0; d < 2; ++d)
        launch_gemm(stream, X0, 256, wih0_bf[d], 256, e_bih0 + d * 768, 768,
                    gi[d], 3200, 768, 256, BIGMOD, 768, 0);
    for (int i = 0; i < 50; ++i) {
        int ts[2] = { i, 49 - i };                  // fwd, bwd
        for (int d = 0; d < 2; ++d) {
            int t = ts[d];
            launch_gemm(stream, hbf_buf + (size_t)d * 64 * 256, 256, whh0_bf[d], 256,
                        e_bhh0 + d * 768, 768, gh[d], 64, 768, 256, BIGMOD, 768, 0);
            fuse(gi[d] + (size_t)t * 64 * 768, 768, gh[d], 768,
                 hf32 + (size_t)d * 64 * 256, 256,
                 hbf_buf + (size_t)d * 64 * 256, 256,
                 xs1 + (size_t)t * 64 * 512 + d * 256, 512, nullptr, 0, 256);
        }
    }

    // layer 1 (input = xs1, K=512)
    hipMemsetAsync(hbf_buf, 0, (size_t)2 * 64 * 256 * 2, stream);
    for (int d = 0; d < 2; ++d)
        launch_gemm(stream, xs1, 512, wih1_bf[d], 512, e_bih1 + d * 768, 768,
                    gi[d], 3200, 768, 512, BIGMOD, 768, 0);
    for (int i = 0; i < 50; ++i) {
        int ts[2] = { i, 49 - i };
        for (int d = 0; d < 2; ++d) {
            int t = ts[d];
            launch_gemm(stream, hbf_buf + (size_t)d * 64 * 256, 256, whh1_bf[d], 256,
                        e_bhh1 + d * 768, 768, gh[d], 64, 768, 256, BIGMOD, 768, 0);
            fuse(gi[d] + (size_t)t * 64 * 768, 768, gh[d], 768,
                 hf32 + (size_t)(2 + d) * 64 * 256, 256,
                 hbf_buf + (size_t)d * 64 * 256, 256,
                 encout_bf + (size_t)t * 64 * 512 + d * 256, 512,
                 encoutf + (size_t)t * 64 * 512 + d * 256, 512, 256);
        }
    }

    // hidden-state projection + attention key precompute
    build_pre<<<(128 * 512) / 256, 256, 0, stream>>>(
        hf32, hf32 + 64 * 256, hf32 + 2 * 64 * 256, hf32 + 3 * 64 * 256, pre_bf);
    launch_gemm(stream, pre_bf, 512, oenc_bf, 512, zbias, 320, hidf,
                128, 320, 512, BIGMOD, 320, 0);
    launch_gemm(stream, encout_bf, 512, eattw_bf, 512, zbias, 320, encatt,
                3200, 320, 512, BIGMOD, 320, 0);
    cvt(hidf, 128, 320, 320, hid_bf, 128, 320);

    // ---- decoder scan (teacher-forced, 32 steps) ----
    for (int t = 0; t < 32; ++t) {
        const ushort_t* h1prev_bf = (t == 0) ? (hid_bf + (size_t)64 * 320)
                                             : (H1all + (size_t)(t - 1) * 64 * 320);
        const ushort_t* h0prev_bf = (t == 0) ? hid_bf : h0bf;

        launch_gemm(stream, h1prev_bf, 320, hattw_bf, 320, hid_att_b, 300,
                    hidatt, 64, 320, 320, BIGMOD, 320, 0);
        attn_score<<<400, 256, 0, stream>>>(encatt, hidatt, Vvec, scores);
        softmax_ctx_x<<<64, 256, 0, stream>>>(scores, encoutf, emb_dec_w, targ, t, xbf);

        launch_gemm(stream, xbf, 832, dwih0_bf, 832, db_ih0, 900,
                    gid, 64, 960, 832, BIGMOD, 960, 0);
        launch_gemm(stream, h0prev_bf, 320, dwhh0_bf, 320, db_hh0, 900,
                    ghd, 64, 960, 320, BIGMOD, 960, 0);
        fuse(gid, 960, ghd, 960, hidf, 320, h0bf, 320, nullptr, 0, nullptr, 0, 300);

        launch_gemm(stream, h0bf, 320, dwih1_bf, 320, db_ih1, 900,
                    gid, 64, 960, 320, BIGMOD, 960, 0);
        launch_gemm(stream, h1prev_bf, 320, dwhh1_bf, 320, db_hh1, 900,
                    ghd, 64, 960, 320, BIGMOD, 960, 0);
        fuse(gid, 960, ghd, 960, hidf + (size_t)64 * 320, 320,
             H1all + (size_t)t * 64 * 320, 320, nullptr, 0, nullptr, 0, 300);
    }

    // ---- batched vocabulary projection: 2048 x 32000 x 320 (2/3 of all FLOPs)
    // row r = t*64+b scatters to out[b*32*32000 + t*32000 + v]
    launch_gemm(stream, H1all, 320, outw_bf, 320, out_b, 32000,
                out, 2048, 32000, 320, 64, 32LL * 32000, 32000LL);
}
